// SPADEResnetBlock_70428873720378
// MI455X (gfx1250) — matrix operand order repaired and checked by reading, not compiled, not run on hardware
//
#include <hip/hip_runtime.h>
#include <stdint.h>

// ---------------------------------------------------------------------------
// SPADE/SEAN ResNet block for MI455X (gfx1250), wave32 + v_wmma_f32_16x16x32_bf16.
// Convs are implicit-GEMM over 9 taps: M=Cout, N=B*H*W=8192, K=Cin.
// Activations: bf16 NHWC. Weights: bf16 [tap][co][ci]. Accum: fp32.
// OOB halo lanes read from a zeroed scratch page -> branch-free inner loop.
// ---------------------------------------------------------------------------

typedef __attribute__((ext_vector_type(16))) __bf16 v16bf;
typedef __attribute__((ext_vector_type(8)))  float  v8f;
typedef __attribute__((ext_vector_type(4)))  int    i32x4;
typedef __attribute__((ext_vector_type(8)))  int    i32x8;

static __device__ __forceinline__ unsigned short f2bf(float f) {
  unsigned u = __builtin_bit_cast(unsigned, f);
  u += 0x7fffu + ((u >> 16) & 1u);          // round-to-nearest-even
  return (unsigned short)(u >> 16);
}

// ---------------- zero the OOB page ----------------
__global__ void k_zero(unsigned short* __restrict__ p, int n) {
  int i = blockIdx.x * blockDim.x + threadIdx.x;
  if (i < n) p[i] = 0;
}

// ---------------- label map from one-hot seg ----------------
__global__ void k_label(const float* __restrict__ seg, int* __restrict__ lbl) {
  int n = blockIdx.x * blockDim.x + threadIdx.x;
  if (n >= 8192) return;
  int b = n >> 12, hw = n & 4095;
  int j = 0;
  for (int c = 0; c < 19; ++c)
    if (seg[(((size_t)(b * 19 + c)) << 12) + hw] > 0.5f) j = c;
  lbl[n] = j;
}

// ---------------- seg -> bf16 NHWC padded to 32 ch ----------------
__global__ void k_segpad(const float* __restrict__ seg, unsigned short* __restrict__ dst) {
  int i = blockIdx.x * blockDim.x + threadIdx.x;
  if (i >= 2 * 4096 * 32) return;
  int c = i & 31, pix = i >> 5;
  int b = pix >> 12, hw = pix & 4095;
  float v = (c < 19) ? seg[(((size_t)(b * 19 + c)) << 12) + hw] : 0.f;
  dst[i] = f2bf(v);
}

// ---------------- mu[b,j,t] = relu(ps_w[j,t,:] . style[b,j,:] + ps_b[j,t]) ----------------
__global__ void k_mu(const float* __restrict__ style, const float* __restrict__ psw,
                     const float* __restrict__ psb, float* __restrict__ mu) {
  int o = blockIdx.x * blockDim.x + threadIdx.x;
  if (o >= 2 * 19 * 512) return;
  int t = o & 511, j = (o >> 9) % 19, b = o / (19 * 512);
  const float* wr = psw + ((size_t)j * 512 + t) * 512;
  const float* sr = style + ((size_t)b * 19 + j) * 512;
  float acc = psb[j * 512 + t];
  for (int s = 0; s < 512; ++s) acc += wr[s] * sr[s];
  mu[o] = fmaxf(acc, 0.f);
}

// ---------------- middle_avg gather -> bf16 NHWC (C=512) ----------------
__global__ void k_mid(const float* __restrict__ mu, const int* __restrict__ lbl,
                      unsigned short* __restrict__ mid) {
  int pix = blockIdx.x;                       // 8192 pixels
  int b = pix >> 12;
  int j = lbl[pix];
  const float* src = mu + ((size_t)b * 19 + j) * 512;
  unsigned short* d = mid + (size_t)pix * 512;
  for (int t = threadIdx.x; t < 512; t += blockDim.x) d[t] = f2bf(src[t]);
}

// ---------------- per-channel mean/rstd of (x + noise*nv) ----------------
__global__ void k_bnstats(const float* __restrict__ x, const float* __restrict__ noise,
                          const float* __restrict__ nv, float* __restrict__ mean,
                          float* __restrict__ rstd, int fin) {
  int c = blockIdx.x;
  float nvc = nv[c];
  float s = 0.f, s2 = 0.f;
  for (int i = threadIdx.x; i < 8192; i += blockDim.x) {
    int b = i >> 12, hw = i & 4095;
    float v = x[(((size_t)(b * fin + c)) << 12) + hw] +
              noise[(b << 12) + ((hw & 63) << 6) + (hw >> 6)] * nvc;
    s += v; s2 += v * v;
  }
  __shared__ float rs[256], rq[256];
  rs[threadIdx.x] = s; rq[threadIdx.x] = s2;
  __syncthreads();
  for (int off = 128; off > 0; off >>= 1) {
    if ((int)threadIdx.x < off) { rs[threadIdx.x] += rs[threadIdx.x + off];
                                  rq[threadIdx.x] += rq[threadIdx.x + off]; }
    __syncthreads();
  }
  if (threadIdx.x == 0) {
    float m = rs[0] * (1.f / 8192.f);
    float var = rq[0] * (1.f / 8192.f) - m * m;
    mean[c] = m;
    rstd[c] = rsqrtf(var + 1e-5f);
  }
}

__global__ void k_bnapply(const float* __restrict__ x, const float* __restrict__ noise,
                          const float* __restrict__ nv, const float* __restrict__ mean,
                          const float* __restrict__ rstd, float* __restrict__ xn, int fin) {
  int i = blockIdx.x * blockDim.x + threadIdx.x;
  if (i >= 2 * fin * 4096) return;
  int hw = i & 4095, c = (i >> 12) % fin, b = i / (fin << 12);
  float v = x[i] + noise[(b << 12) + ((hw & 63) << 6) + (hw >> 6)] * nv[c];
  xn[i] = (v - mean[c]) * rstd[c];
}

// ---------------- spectral norm: inv_sigma via one power iteration ----------------
__global__ __launch_bounds__(1024) void k_sigma(const float* __restrict__ w,
                                                const float* __restrict__ u,
                                                int M, int K, float* __restrict__ inv_sigma) {
  __shared__ float sv[4608];
  __shared__ float red[1024];
  float loc = 0.f;
  for (int j = threadIdx.x; j < K; j += 1024) {
    float a = 0.f;
    for (int i = 0; i < M; ++i) a += w[(size_t)i * K + j] * u[i];
    sv[j] = a; loc += a * a;
  }
  red[threadIdx.x] = loc;
  __syncthreads();
  for (int off = 512; off > 0; off >>= 1) {
    if ((int)threadIdx.x < off) red[threadIdx.x] += red[threadIdx.x + off];
    __syncthreads();
  }
  float invn = 1.f / (sqrtf(red[0]) + 1e-12f);
  __syncthreads();
  float loc2 = 0.f;
  for (int i = threadIdx.x; i < M; i += 1024) {
    const float* wr = w + (size_t)i * K;
    float a = 0.f;
    for (int j = 0; j < K; ++j) a += wr[j] * sv[j];
    a *= invn;
    loc2 += a * a;
  }
  red[threadIdx.x] = loc2;
  __syncthreads();
  for (int off = 512; off > 0; off >>= 1) {
    if ((int)threadIdx.x < off) red[threadIdx.x] += red[threadIdx.x + off];
    __syncthreads();
  }
  if (threadIdx.x == 0) {
    float s2 = red[0];
    float sigma = s2 / (sqrtf(s2) + 1e-12f);
    *inv_sigma = 1.f / sigma;
  }
}

// ---------------- weight repack OIHW f32 -> [tap][co][ci] bf16 (K zero-padded, SN scale) ----
__global__ void k_wconv(const float* __restrict__ w, unsigned short* __restrict__ dst,
                        const float* __restrict__ scale_p, int M, int Ksrc, int Kpad, int ntaps) {
  int i = blockIdx.x * blockDim.x + threadIdx.x;
  if (i >= ntaps * M * Kpad) return;
  int ci = i % Kpad, r = i / Kpad;
  int co = r % M, tap = r / M;
  float s = scale_p ? *scale_p : 1.f;
  float v = (ci < Ksrc) ? w[((size_t)co * Ksrc + ci) * ntaps + tap] * s : 0.f;
  dst[i] = f2bf(v);
}

// ---------------- WMMA implicit-GEMM conv (3x3 pad=1, or 1x1) ----------------
// X: bf16 NHWC (C=K).  Wt: bf16 [tap][M][K].  ZP: >=2KB of zeros for halo lanes.
// Block = 8 waves (2M x 4N); block tile 64(M) x 256(N); wave tile 32(M) x 64(N).
// ep==0: Of32 NCHW = alpha*(conv+bias) [+skip] [+prev];  ep==1: Ob16 NHWC = bf16(relu(conv+bias))
__global__ __launch_bounds__(256) void k_conv(
    const unsigned short* __restrict__ X, const unsigned short* __restrict__ Wt,
    const unsigned short* __restrict__ ZP,
    const float* __restrict__ bias, const float* __restrict__ gate, int gate_mode,
    float* __restrict__ Of32, unsigned short* __restrict__ Ob16,
    const float* __restrict__ skip, int accum, int M, int K, int ntaps, int ep) {
  int lane = threadIdx.x & 31;
  int wv   = threadIdx.x >> 5;                 // 8 waves: 2(M) x 4(N)
  int m_wave = blockIdx.x * 64 + (wv & 1) * 32;
  int n_wave = blockIdx.y * 256 + (wv >> 1) * 64;
  int lm    = lane & 15;
  int khalf = (lane >> 4) * 8;                 // A: hi lanes take K-half offset 8
  int koff  = (lane >> 4) * 8;                 // B: hi lanes take K-half offset 8

  v8f acc[2][4];
#pragma unroll
  for (int a = 0; a < 2; ++a)
#pragma unroll
    for (int b = 0; b < 4; ++b)
#pragma unroll
      for (int r = 0; r < 8; ++r) acc[a][b][r] = 0.f;

  // per-lane pixel coordinates for the 4 N-fragments (k-invariant)
  int pb[4], ph[4], pw[4];
#pragma unroll
  for (int nf = 0; nf < 4; ++nf) {
    int n = n_wave + nf * 16 + lm;
    pb[nf] = n >> 12;
    ph[nf] = (n & 4095) >> 6;
    pw[nf] = n & 63;
  }

  for (int tap = 0; tap < ntaps; ++tap) {
    int dh = (ntaps == 9) ? tap / 3 - 1 : 0;
    int dw = (ntaps == 9) ? tap % 3 - 1 : 0;
    const unsigned short* wtap = Wt + (size_t)tap * M * K;
    const unsigned short* wbaseA = wtap + (size_t)(m_wave + lm) * K + khalf;

    // branch-free: invalid halo lanes read zeros from ZP
    const unsigned short* bbase[4];
#pragma unroll
    for (int nf = 0; nf < 4; ++nf) {
      int ih = ph[nf] + dh, iw = pw[nf] + dw;
      bool ok = ((unsigned)ih < 64u) & ((unsigned)iw < 64u);
      bbase[nf] = ok ? (X + ((size_t)((pb[nf] << 12) + (ih << 6) + iw)) * K + koff)
                     : (ZP + koff);
    }

    for (int k0 = 0; k0 < K; k0 += 32) {
      i32x8 afr[2], bfr[4];
#pragma unroll
      for (int mf = 0; mf < 2; ++mf) {
        const unsigned short* p = wbaseA + (size_t)mf * 16 * K + k0;
        i32x4 lo = *(const i32x4*)p;           // K = khalf .. khalf+7
        i32x4 hi = *(const i32x4*)(p + 16);    // K = 16+khalf .. 16+khalf+7
        afr[mf][0] = lo[0]; afr[mf][1] = lo[1]; afr[mf][2] = lo[2]; afr[mf][3] = lo[3];
        afr[mf][4] = hi[0]; afr[mf][5] = hi[1]; afr[mf][6] = hi[2]; afr[mf][7] = hi[3];
      }
      if (k0 + 32 < K)                         // gfx1250 global_prefetch path
        __builtin_prefetch(wbaseA + k0 + 32, 0, 1);
#pragma unroll
      for (int nf = 0; nf < 4; ++nf) {
        const unsigned short* p = bbase[nf] + k0;
        i32x4 lo = *(const i32x4*)p;           // K = koff .. koff+7
        i32x4 hi = *(const i32x4*)(p + 16);    // K = 16+koff .. 16+koff+7
        bfr[nf][0] = lo[0]; bfr[nf][1] = lo[1]; bfr[nf][2] = lo[2]; bfr[nf][3] = lo[3];
        bfr[nf][4] = hi[0]; bfr[nf][5] = hi[1]; bfr[nf][6] = hi[2]; bfr[nf][7] = hi[3];
      }
#pragma unroll
      for (int mf = 0; mf < 2; ++mf)
#pragma unroll
        for (int nf = 0; nf < 4; ++nf)
          acc[mf][nf] = __builtin_amdgcn_wmma_f32_16x16x32_bf16(
              false, __builtin_bit_cast(v16bf, afr[mf]),
              false, __builtin_bit_cast(v16bf, bfr[nf]),
              (short)0, acc[mf][nf], false, false);
    }
  }

  float alpha = 1.f;
  if (gate_mode != 0) {
    float g = *gate;
    float sg = 1.f / (1.f + expf(-g));
    alpha = (gate_mode == 2) ? sg : 1.f - sg;
  }
#pragma unroll
  for (int mf = 0; mf < 2; ++mf) {
#pragma unroll
    for (int nf = 0; nf < 4; ++nf) {
      int n = n_wave + nf * 16 + lm;
      int b = n >> 12, hw = n & 4095;
      int mbase = m_wave + mf * 16 + (lane >> 4) * 8;   // C layout: hi lanes hold M+8
#pragma unroll
      for (int r = 0; r < 8; ++r) {
        int m = mbase + r;
        float v = acc[mf][nf][r] + (bias ? bias[m] : 0.f);
        if (ep == 1) {
          Ob16[(size_t)n * M + m] = f2bf(fmaxf(v, 0.f));
        } else {
          size_t idx = (((size_t)b * M + m) << 12) + hw;
          float o = alpha * v;
          if (skip)  o += skip[idx];
          if (accum) o += Of32[idx];
          Of32[idx] = o;
        }
      }
    }
  }
}

// ---------------- ACE output fuse: xn*(1+g)+b [, leaky] -> bf16 NHWC ----------------
__global__ void k_acefuse(const float* __restrict__ xn, const float* __restrict__ g,
                          const float* __restrict__ bt, unsigned short* __restrict__ out,
                          int fin, int do_leaky) {
  int i = blockIdx.x * blockDim.x + threadIdx.x;
  if (i >= 2 * fin * 4096) return;
  int hw = i & 4095, c = (i >> 12) % fin, b = i / (fin << 12);
  float v = xn[i] * (1.f + g[i]) + bt[i];
  if (do_leaky) v = (v >= 0.f) ? v : 0.2f * v;
  out[((size_t)((b << 12) + hw)) * fin + c] = f2bf(v);
}

// ---------------------------------------------------------------------------
extern "C" void kernel_launch(void* const* d_in, const int* in_sizes, int n_in,
                              void* d_out, int out_size, void* d_ws, size_t ws_size,
                              hipStream_t stream) {
  (void)in_sizes; (void)n_in; (void)out_size; (void)ws_size;
  auto in = [&](int i) { return (const float*)d_in[i]; };

  const float* X   = in(0);
  const float* SEG = in(1);
  const float* STY = in(2);
  const float* NZ[3] = { in(3), in(4), in(5) };      // noise_s, noise_0, noise_1
  // params flattened in insertion order: ace_s(15), ace_0(15), ace_1(15), then convs
  const int AB[3] = { 6, 21, 36 };
  const float* CONV0_W = in(51);
  const float* CONV0_B = in(52);
  const float* CONV1_W = in(53);
  const float* CONV1_B = in(54);
  const float* CONVS_W = in(55);
  const float* U0 = in(56);
  const float* U1 = in(57);
  const float* US = in(58);

  char* ws = (char*)d_ws;
  size_t off = 0;
  auto alloc = [&](size_t bytes) {
    size_t r = off; off += (bytes + 255) & ~(size_t)255; return r;
  };
  int*            LBL   = (int*)(ws + alloc(8192 * 4));
  float*          MEAN  = (float*)(ws + alloc(512 * 4));
  float*          RSTD  = (float*)(ws + alloc(512 * 4));
  float*          INV   = (float*)(ws + alloc(64));
  unsigned short* ZP    = (unsigned short*)(ws + alloc(4096));       // zero page
  float*          MU    = (float*)(ws + alloc(2 * 19 * 512 * 4));
  unsigned short* SEG16 = (unsigned short*)(ws + alloc((size_t)2 * 4096 * 32 * 2));
  unsigned short* ACT16 = (unsigned short*)(ws + alloc((size_t)2 * 4096 * 128 * 2));
  unsigned short* MID16 = (unsigned short*)(ws + alloc((size_t)2 * 4096 * 512 * 2));
  unsigned short* XA16  = (unsigned short*)(ws + alloc((size_t)2 * 4096 * 512 * 2));
  unsigned short* WB    = (unsigned short*)(ws + alloc((size_t)9 * 512 * 512 * 2));
  float*          XN    = (float*)(ws + alloc((size_t)2 * 512 * 4096 * 4));
  float*          GAM   = (float*)(ws + alloc((size_t)2 * 512 * 4096 * 4));
  float*          BET   = (float*)(ws + alloc((size_t)2 * 512 * 4096 * 4));
  float*          XS    = (float*)(ws + alloc((size_t)2 * 256 * 4096 * 4));
  float*          HB    = (float*)(ws + alloc((size_t)2 * 256 * 4096 * 4));

  auto conv = [&](const unsigned short* Xp, const unsigned short* Wp, const float* bias,
                  const float* gate, int gmode, float* Of, unsigned short* Ob,
                  const float* skip, int accum, int M, int K, int ntaps, int ep) {
    dim3 g(M / 64, 32);
    k_conv<<<g, 256, 0, stream>>>(Xp, Wp, ZP, bias, gate, gmode, Of, Ob, skip, accum,
                                  M, K, ntaps, ep);
  };
  auto wconv = [&](const float* w, const float* sc, int M, int Ksrc, int Kpad, int nt) {
    int tot = nt * M * Kpad;
    k_wconv<<<(tot + 255) / 256, 256, 0, stream>>>(w, WB, sc, M, Ksrc, Kpad, nt);
  };

  k_zero<<<8, 256, 0, stream>>>(ZP, 2048);
  k_label<<<32, 256, 0, stream>>>(SEG, LBL);
  k_segpad<<<(2 * 4096 * 32 + 255) / 256, 256, 0, stream>>>(SEG, SEG16);
  k_sigma<<<1, 1024, 0, stream>>>(CONVS_W, US, 256, 512,  INV + 0);
  k_sigma<<<1, 1024, 0, stream>>>(CONV0_W, U0, 256, 4608, INV + 1);
  k_sigma<<<1, 1024, 0, stream>>>(CONV1_W, U1, 256, 2304, INV + 2);

  for (int a = 0; a < 3; ++a) {           // 0: ace_s, 1: ace_0, 2: ace_1
    int base = AB[a];
    int fin = (a == 2) ? 256 : 512;
    const float* nv   = in(base + 0);
    const float* bg   = in(base + 1);
    const float* bb   = in(base + 2);
    const float* spw0 = in(base + 3);
    const float* spb0 = in(base + 4);
    const float* spwg = in(base + 5);
    const float* spbg = in(base + 6);
    const float* spwb = in(base + 7);
    const float* spbb = in(base + 8);
    const float* psw  = in(base + 9);
    const float* psb  = in(base + 10);
    const float* cgw  = in(base + 11);
    const float* cgb  = in(base + 12);
    const float* cbw  = in(base + 13);
    const float* cbb  = in(base + 14);
    const float* noise = NZ[a];
    const float* xsrc  = (a == 2) ? HB : X;
    int tot = 2 * fin * 4096;

    k_mu<<<(2 * 19 * 512 + 255) / 256, 256, 0, stream>>>(STY, psw, psb, MU);
    k_mid<<<8192, 128, 0, stream>>>(MU, LBL, MID16);
    k_bnstats<<<fin, 256, 0, stream>>>(xsrc, noise, nv, MEAN, RSTD, fin);
    k_bnapply<<<(tot + 255) / 256, 256, 0, stream>>>(xsrc, noise, nv, MEAN, RSTD, XN, fin);

    // SPADE hidden: 19(->32 pad) -> 128, relu, bf16 NHWC
    wconv(spw0, nullptr, 128, 19, 32, 9);
    conv(SEG16, WB, spb0, nullptr, 0, nullptr, ACT16, nullptr, 0, 128, 32, 9, 1);
    // SPADE gamma/beta: scaled by (1 - sigmoid(gate))
    wconv(spwg, nullptr, fin, 128, 128, 9);
    conv(ACT16, WB, spbg, bg, 1, GAM, nullptr, nullptr, 0, fin, 128, 9, 0);
    wconv(spwb, nullptr, fin, 128, 128, 9);
    conv(ACT16, WB, spbb, bb, 1, BET, nullptr, nullptr, 0, fin, 128, 9, 0);
    // Style gamma/beta: accumulate sigmoid(gate) * conv
    wconv(cgw, nullptr, fin, 512, 512, 9);
    conv(MID16, WB, cgb, bg, 2, GAM, nullptr, nullptr, 1, fin, 512, 9, 0);
    wconv(cbw, nullptr, fin, 512, 512, 9);
    conv(MID16, WB, cbb, bb, 2, BET, nullptr, nullptr, 1, fin, 512, 9, 0);
    // ACE out (+leaky on main path) -> bf16 NHWC
    k_acefuse<<<(tot + 255) / 256, 256, 0, stream>>>(XN, GAM, BET, XA16, fin, a != 0);

    if (a == 0) {        // conv_s (1x1, SN, no bias) -> XS
      wconv(CONVS_W, INV + 0, 256, 512, 512, 1);
      conv(XA16, WB, nullptr, nullptr, 0, XS, nullptr, nullptr, 0, 256, 512, 1, 0);
    } else if (a == 1) { // conv_0 (3x3, SN) -> HB
      wconv(CONV0_W, INV + 1, 256, 512, 512, 9);
      conv(XA16, WB, CONV0_B, nullptr, 0, HB, nullptr, nullptr, 0, 256, 512, 9, 0);
    } else {             // conv_1 (3x3, SN) + skip XS -> d_out
      wconv(CONV1_W, INV + 2, 256, 256, 256, 9);
      conv(XA16, WB, CONV1_B, nullptr, 0, (float*)d_out, nullptr, XS, 0, 256, 256, 9, 0);
    }
  }
}
